// HLSGNN_84920093377237
// MI455X (gfx1250) — matrix operand order repaired and checked by reading, not compiled, not run on hardware
//
#include <hip/hip_runtime.h>
#include <math.h>

// ---------------------------------------------------------------------------
// HGT conv for MI455X (gfx1250, wave32). Dense parts use V_WMMA_F32_16X16X32_F16.
// ---------------------------------------------------------------------------

typedef __attribute__((ext_vector_type(16))) _Float16 v16h;
typedef __attribute__((ext_vector_type(8)))  float    v8f;

// Monotonic float<->uint encoding so atomicMax(uint) == float max.
__device__ __forceinline__ unsigned fkey(float f) {
    unsigned u = __float_as_uint(f);
    return (u & 0x80000000u) ? ~u : (u | 0x80000000u);
}
__device__ __forceinline__ float fdec(unsigned k) {
    unsigned u = (k & 0x80000000u) ? (k & 0x7fffffffu) : ~k;
    return __uint_as_float(u);
}

// ---------------------------------------------------------------------------
// Compose W_eff = Wsrc[:, coff:coff+64] x blockdiag(rel[h]) ; b_eff likewise.
// Wsrc: (64,192) row-major, rel: (4,16,16), Weff: (64,64), beff: (64)
// ---------------------------------------------------------------------------
__global__ void compose_rel(const float* __restrict__ Wsrc,
                            const float* __restrict__ bsrc,
                            const float* __restrict__ rel,
                            int coff,
                            float* __restrict__ Weff,
                            float* __restrict__ beff) {
    int t = threadIdx.x;                 // 256 threads
    for (int idx = t; idx < 64 * 64; idx += 256) {
        int i = idx >> 6, c = idx & 63;
        int h = c >> 4, e = c & 15;
        float s = 0.f;
        #pragma unroll
        for (int d = 0; d < 16; ++d)
            s += Wsrc[i * 192 + coff + h * 16 + d] * rel[(h * 16 + d) * 16 + e];
        Weff[idx] = s;
    }
    if (t < 64) {
        int h = t >> 4, e = t & 15;
        float s = 0.f;
        #pragma unroll
        for (int d = 0; d < 16; ++d)
            s += bsrc[coff + h * 16 + d] * rel[(h * 16 + d) * 16 + e];
        beff[t] = s;
    }
}

// ---------------------------------------------------------------------------
// out(M x 64) = f16(A(M x 64)) @ f16(W[k][coff+n]) + bias[n]
// 256 threads = 8 waves/block; each wave owns one 16-row strip and loops over
// the 4 column tiles; two v_wmma_f32_16x16x32_f16 per tile (K = 64).
// A fragments loaded as float4 (global_load_b128), OOB rows zeroed via mask.
// If skip != nullptr: out = g*val + (1-g)*xres with g = sigmoid(*skip).
// A/B/C register layouts per CDNA5 ISA 7.12.2 (wave32).
// ---------------------------------------------------------------------------
__global__ void __launch_bounds__(256)
wmma_gemm64(const float* __restrict__ A, int M,
            const float* __restrict__ W, int ldw, int coff,
            const float* __restrict__ bias,
            float* __restrict__ out,
            const float* __restrict__ skip,
            const float* __restrict__ xres) {
    const int wave    = threadIdx.x >> 5;
    const int strip   = blockIdx.x * 8 + wave;
    const int nstrips = (M + 15) >> 4;
    if (strip >= nstrips) return;            // wave-uniform: EXEC stays all-ones

    const int lane  = threadIdx.x & 31;
    const int arow  = strip * 16 + (lane & 15);
    const bool rowok = (arow < M);
    const float msk  = rowok ? 1.f : 0.f;
    const int arow_c = rowok ? arow : 0;
    const int kbA = (lane & 16) ? 8 : 0;     // A half-K base per lane group
    const int q0  = kbA >> 2;                // float4 index offset (0 or 2)

    const float4* A4 = reinterpret_cast<const float4*>(A + (size_t)arow_c * 64);
    float4 p[8];
    #pragma unroll
    for (int ch = 0; ch < 4; ++ch) {         // 4 chunks of 8 floats (2 x b128)
        p[2 * ch]     = A4[ch * 4 + q0];
        p[2 * ch + 1] = A4[ch * 4 + q0 + 1];
    }
    v16h a0, a1;
    #pragma unroll
    for (int ch = 0; ch < 2; ++ch) {         // a0 halves: chunks 0,1 (K 0..31)
        const float4 u = p[2 * ch], w = p[2 * ch + 1];
        a0[8 * ch + 0] = (_Float16)(u.x * msk);
        a0[8 * ch + 1] = (_Float16)(u.y * msk);
        a0[8 * ch + 2] = (_Float16)(u.z * msk);
        a0[8 * ch + 3] = (_Float16)(u.w * msk);
        a0[8 * ch + 4] = (_Float16)(w.x * msk);
        a0[8 * ch + 5] = (_Float16)(w.y * msk);
        a0[8 * ch + 6] = (_Float16)(w.z * msk);
        a0[8 * ch + 7] = (_Float16)(w.w * msk);
    }
    #pragma unroll
    for (int ch = 0; ch < 2; ++ch) {         // a1 halves: chunks 2,3 (K 32..63)
        const float4 u = p[4 + 2 * ch], w = p[4 + 2 * ch + 1];
        a1[8 * ch + 0] = (_Float16)(u.x * msk);
        a1[8 * ch + 1] = (_Float16)(u.y * msk);
        a1[8 * ch + 2] = (_Float16)(u.z * msk);
        a1[8 * ch + 3] = (_Float16)(u.w * msk);
        a1[8 * ch + 4] = (_Float16)(w.x * msk);
        a1[8 * ch + 5] = (_Float16)(w.y * msk);
        a1[8 * ch + 6] = (_Float16)(w.z * msk);
        a1[8 * ch + 7] = (_Float16)(w.w * msk);
    }

    float g = 1.f;
    const bool gated = (skip != nullptr);
    if (gated) g = 1.f / (1.f + expf(-skip[0]));

    const int kbB  = (lane & 16) ? 8 : 0;    // B half-K base per lane group
    const int ncol = lane & 15;
    const int rofs = (lane & 16) ? 8 : 0;
    const bool fullstrip = (strip * 16 + 16) <= M;   // wave-uniform

    for (int nt = 0; nt < 4; ++nt) {
        const int col = nt * 16 + ncol;
        v16h b0, b1;
        #pragma unroll
        for (int j = 0; j < 16; ++j) {
            b0[j] = (_Float16)W[(size_t)(kbB + (j & 7) + ((j & 8) << 1)) * ldw + coff + col];
            b1[j] = (_Float16)W[(size_t)(32 + kbB + (j & 7) + ((j & 8) << 1)) * ldw + coff + col];
        }
        v8f c = {};
        c = __builtin_amdgcn_wmma_f32_16x16x32_f16(false, a0, false, b0,
                                                   (short)0, c, false, false);
        c = __builtin_amdgcn_wmma_f32_16x16x32_f16(false, a1, false, b1,
                                                   (short)0, c, false, false);
        const float bv = bias[col];
        float* op = out + (size_t)(strip * 16 + rofs) * 64 + col;
        if (fullstrip) {
            if (gated) {
                const float* xp = xres + (size_t)(strip * 16 + rofs) * 64 + col;
                #pragma unroll
                for (int r = 0; r < 8; ++r)
                    op[(size_t)r * 64] = g * (c[r] + bv) +
                                         (1.f - g) * xp[(size_t)r * 64];
            } else {
                #pragma unroll
                for (int r = 0; r < 8; ++r)
                    op[(size_t)r * 64] = c[r] + bv;
            }
        } else {
            #pragma unroll
            for (int r = 0; r < 8; ++r) {
                int orow = strip * 16 + r + rofs;
                if (orow < M) {
                    float v = c[r] + bv;
                    if (gated) v = g * v + (1.f - g) * xres[(size_t)orow * 64 + col];
                    out[(size_t)orow * 64 + col] = v;
                }
            }
        }
    }
}

// ---------------------------------------------------------------------------
// Pass E1: per (edge, head) score s = <q[dst,h], kk[src,h]> * p_rel[h] * scale
//          store s; atomicMax running per-(dst,head) max (uint-encoded).
// ---------------------------------------------------------------------------
__global__ void edge_score(const int* __restrict__ src, const int* __restrict__ dst,
                           int E,
                           const float* __restrict__ q,
                           const float* __restrict__ kk,
                           const float* __restrict__ prel, float scale,
                           float* __restrict__ sbuf,
                           unsigned* __restrict__ mkey) {
    int t = blockIdx.x * blockDim.x + threadIdx.x;
    if (t >= E * 4) return;
    int e = t >> 2, h = t & 3;
    int s = src[e], d = dst[e];
    const float* qp = q  + (size_t)d * 64 + h * 16;
    const float* kp = kk + (size_t)s * 64 + h * 16;
    float acc = 0.f;
    #pragma unroll
    for (int i = 0; i < 16; ++i) acc += qp[i] * kp[i];
    acc *= prel[h] * scale;
    sbuf[t] = acc;
    atomicMax(&mkey[d * 4 + h], fkey(acc));
}

// ---------------------------------------------------------------------------
// Pass E2: e = exp(s - m[dst,h]); z[dst,h] += e; agg[dst,h,:] += e * vv[src,h,:]
// (division by z deferred to the node epilogue: sum(e*v)/z == sum(alpha*v))
// ---------------------------------------------------------------------------
__global__ void edge_agg(const int* __restrict__ src, const int* __restrict__ dst,
                         int E,
                         const float* __restrict__ sbuf,
                         const unsigned* __restrict__ mkey,
                         const float* __restrict__ vv,
                         float* __restrict__ z,
                         float* __restrict__ agg) {
    int t = blockIdx.x * blockDim.x + threadIdx.x;
    if (t >= E * 4) return;
    int e = t >> 2, h = t & 3;
    int s = src[e], d = dst[e];
    float m = fdec(mkey[d * 4 + h]);
    float ev = expf(sbuf[t] - m);
    atomicAdd(&z[d * 4 + h], ev);
    const float* vp = vv  + (size_t)s * 64 + h * 16;
    float*       ap = agg + (size_t)d * 64 + h * 16;
    #pragma unroll
    for (int i = 0; i < 16; ++i) atomicAdd(&ap[i], ev * vp[i]);
}

// ---------------------------------------------------------------------------
// Node epilogue: agg <- gelu_exact( agg / (z + 1e-16) )   (in place)
// ---------------------------------------------------------------------------
__global__ void node_gelu(float* __restrict__ agg, const float* __restrict__ z,
                          int n) {
    int t = blockIdx.x * blockDim.x + threadIdx.x;
    if (t >= n * 64) return;
    int node = t >> 6, c = t & 63, h = c >> 4;
    float a = agg[t] / (z[node * 4 + h] + 1e-16f);
    agg[t] = 0.5f * a * (1.f + erff(a * 0.7071067811865475f));
}

// ---------------------------------------------------------------------------
// Host-side orchestration.
// Input leaf ordering: JAX pytree (dict keys sorted alphabetically, all levels).
// ---------------------------------------------------------------------------
struct KeyD { int e, x, w, b, kr, vr, pr; };
struct SubD { int nk; KeyD k[4]; int qw, qb, ow, ob, sk, slot; };

static const SubD SUBS[5] = {
    // ap: 1 edge type into instr: array__mem__instr
    {1, {{0, 217, 19, 16, 13, 34, 28}}, 21, 18, 27, 24, 33, 0},
    // unroll: block__hrchy__instr
    {1, {{2, 219, 191, 187, 182, 212, 203}}, 192, 188, 200, 196, 209, 1},
    // pipeline: array__mem, block__hrchy, instr__data, instr__mem -> instr
    {4, {{0, 217, 144, 139, 130, 173, 159},
         {2, 219, 145, 140, 131, 174, 160},
         {4, 220, 146, 141, 133, 176, 162},
         {7, 220, 146, 141, 136, 179, 165}}, 146, 141, 156, 151, 170, 2},
    // merge: same edge structure as pipeline
    {4, {{0, 217, 92, 87, 78, 121, 107},
         {2, 219, 93, 88, 79, 122, 108},
         {4, 220, 94, 89, 81, 124, 110},
         {7, 220, 94, 89, 84, 127, 113}}, 94, 89, 104, 99, 118, 3},
    // flatten: block__hrchy, instr__data, instr__mem -> instr
    {3, {{2, 219, 48, 44, 37, 71, 60},
         {4, 220, 49, 45, 39, 73, 62},
         {7, 220, 49, 45, 41, 75, 64}}, 49, 45, 57, 53, 68, 4},
};

extern "C" void kernel_launch(void* const* d_in, const int* in_sizes, int n_in,
                              void* d_out, int out_size, void* d_ws, size_t ws_size,
                              hipStream_t stream) {
    (void)n_in; (void)out_size;
    const size_t NI = 100000;            // instr node count

    float* Wsp = (float*)d_ws;
    float*    q    = Wsp;                          // 6,400,000
    unsigned* mk   = (unsigned*)(Wsp + 6400000);   //   400,000
    float*    z    = Wsp + 6800000;                //   400,000
    float*    agg  = Wsp + 7200000;                // 6,400,000
    float*    sb   = Wsp + 13600000;               // 2,120,000 (max 530k edges * 4)
    float*    kk   = Wsp + 15720000;               // 14,400,000 (max 225k rows * 64)
    float*    vv   = Wsp + 30120000;               // 14,400,000
    float*    weff = Wsp + 44520000;               //     4,096
    float*    beff = Wsp + 44524096;               //        64
    // total ~44.53M floats (~179 MB) <= ws_size (assumed)
    (void)ws_size;

    #define FP(i) ((const float*)d_in[(i)])

    const float* x_instr = FP(220);
    const unsigned gemmBlocks = (unsigned)(((NI + 15) / 16 + 7) / 8);

    for (int si = 0; si < 5; ++si) {
        const SubD& S = SUBS[si];

        hipMemsetAsync(mk,  0, NI * 4  * sizeof(unsigned), stream);
        hipMemsetAsync(z,   0, NI * 4  * sizeof(float),    stream);
        hipMemsetAsync(agg, 0, NI * 64 * sizeof(float),    stream);

        // q_instr = x_instr @ kqv_w[:,64:128] + kqv_b[64:128]
        wmma_gemm64<<<dim3(gemmBlocks), dim3(256), 0, stream>>>(
            x_instr, (int)NI, FP(S.qw), 192, 64, FP(S.qb) + 64, q, nullptr, nullptr);

        long ro[4], eo[4];
        long rcur = 0, ecur = 0;
        for (int ki = 0; ki < S.nk; ++ki) {
            const KeyD& K = S.k[ki];
            const int nsrc = in_sizes[K.x] / 64;
            const int E    = in_sizes[K.e] / 2;
            const unsigned gb = (unsigned)((((nsrc + 15) / 16) + 7) / 8);
            ro[ki] = rcur; eo[ki] = ecur;

            // kk = x_src @ (Wk . blockdiag(k_rel)) + bk_eff
            compose_rel<<<dim3(1), dim3(256), 0, stream>>>(
                FP(K.w), FP(K.b), FP(K.kr), 0, weff, beff);
            wmma_gemm64<<<dim3(gb), dim3(256), 0, stream>>>(
                FP(K.x), nsrc, weff, 64, 0, beff, kk + rcur * 64, nullptr, nullptr);

            // vv = x_src @ (Wv . blockdiag(v_rel)) + bv_eff
            compose_rel<<<dim3(1), dim3(256), 0, stream>>>(
                FP(K.w), FP(K.b), FP(K.vr), 128, weff, beff);
            wmma_gemm64<<<dim3(gb), dim3(256), 0, stream>>>(
                FP(K.x), nsrc, weff, 64, 0, beff, vv + rcur * 64, nullptr, nullptr);

            const int* eptr = (const int*)d_in[K.e];
            edge_score<<<dim3((unsigned)((E * 4 + 255) / 256)), dim3(256), 0, stream>>>(
                eptr, eptr + E, E, q, kk + rcur * 64, FP(K.pr), 0.25f,
                sb + ecur * 4, mk);

            rcur += nsrc; ecur += E;
        }

        for (int ki = 0; ki < S.nk; ++ki) {
            const KeyD& K = S.k[ki];
            const int E = in_sizes[K.e] / 2;
            const int* eptr = (const int*)d_in[K.e];
            edge_agg<<<dim3((unsigned)((E * 4 + 255) / 256)), dim3(256), 0, stream>>>(
                eptr, eptr + E, E, sb + eo[ki] * 4, mk, vv + ro[ki] * 64, z, agg);
        }

        node_gelu<<<dim3((unsigned)((NI * 64 + 255) / 256)), dim3(256), 0, stream>>>(
            agg, z, (int)NI);

        // out = g * (gelu(agg/z) @ out_w + out_b) + (1-g) * x_instr
        wmma_gemm64<<<dim3(gemmBlocks), dim3(256), 0, stream>>>(
            agg, (int)NI, FP(S.ow), 64, 0, FP(S.ob),
            (float*)d_out + (size_t)S.slot * NI * 64, FP(S.sk), x_instr);
    }
    #undef FP
}
